// BasicBlock_AresB_63565515981155
// MI455X (gfx1250) — matrix operand order repaired and checked by reading, not compiled, not run on hardware
//
#include <hip/hip_runtime.h>
#include <hip/hip_bf16.h>

// ---------------------------------------------------------------------------
// Binarized ShuffleNet block for MI455X (gfx1250), wave32.
// Convs: implicit GEMM on V_WMMA_F32_16X16X32_F16.
// LDS staging: TDM tensor_load_to_lds for weights, async global->LDS for
// activations (both guarded; sync fallback keeps the file compiling anywhere).
// ---------------------------------------------------------------------------

typedef __attribute__((ext_vector_type(16))) _Float16 v16h;
typedef __attribute__((ext_vector_type(8)))  _Float16 h8;
typedef __attribute__((ext_vector_type(8)))  float    v8f;
typedef __attribute__((ext_vector_type(4))) unsigned int u32x4;
typedef __attribute__((ext_vector_type(8))) int          i32x8;
typedef __attribute__((ext_vector_type(4))) int          i32x4;

#define AS1 __attribute__((address_space(1)))
#define AS3 __attribute__((address_space(3)))

#define BN_   32
#define BH    56
#define BW    56
#define BHW   3136                 // 56*56
#define BNHW  100352               // 32*3136
#define CIO   256
#define CMID  128
#define KDIM  1152                 // 128 * 9
#define KT    32
#define KITER 36                   // 1152/32
#define NTILE 128
#define MTILE 64
#define EPSV  1e-5f
#define SLOPE 0.01f
#define INV_NHW (1.0f / 100352.0f)

// ---- CDNA5 feature gates ---------------------------------------------------
#if __has_builtin(__builtin_amdgcn_tensor_load_to_lds)
#define HAVE_TDM 1
#else
#define HAVE_TDM 0
#endif

#if __has_builtin(__builtin_amdgcn_global_load_async_to_lds_b128)
#define HAVE_ASYNC_LDS 1
#else
#define HAVE_ASYNC_LDS 0
#endif

__device__ __forceinline__ void wait_tensor_zero() {
#if __has_builtin(__builtin_amdgcn_s_wait_tensorcnt)
    __builtin_amdgcn_s_wait_tensorcnt(0);
#else
    asm volatile("s_wait_tensorcnt 0" ::: "memory");
#endif
}

__device__ __forceinline__ void wait_async_zero() {
#if __has_builtin(__builtin_amdgcn_s_wait_asynccnt)
    __builtin_amdgcn_s_wait_asynccnt(0);
#else
    asm volatile("s_wait_asynccnt 0" ::: "memory");
#endif
}

#if HAVE_TDM
// 2D tile DMA: global (row-major, row stride in elements) -> LDS, f16 elems.
// D# layout per CDNA5 ISA ch.8: group0 = {flags, lds_addr, global_addr, type},
// group1 = {data_size, tensor dims, tile dims, dim0 stride}. 2D => groups 2/3
// zero/disabled.
__device__ __forceinline__ void tdm_load_tile_f16(
    unsigned lds_byte_off, const void* gptr,
    int tile_w, int tile_h, int row_stride_elems)
{
    unsigned long long ga = (unsigned long long)gptr;
    u32x4 g0;
    g0[0] = 1u;                                        // count=1 (valid D#)
    g0[1] = lds_byte_off;                              // lds_addr
    g0[2] = (unsigned)(ga & 0xffffffffu);              // global_addr[31:0]
    g0[3] = (unsigned)((ga >> 32) & 0x01ffffffu)       // global_addr[56:32]
          | 0x80000000u;                               // type=2 ("image")
    i32x8 g1;
    g1[0] = 0x00010000;                                // data_size=1 -> 2B
    g1[1] = (tile_w & 0xffff) << 16;                   // tensor_dim0[15:0]
    g1[2] = (tile_h & 0xffff) << 16;                   // tensor_dim1[15:0]
    g1[3] = (tile_w & 0xffff) << 16;                   // tile_dim0
    g1[4] = (tile_h & 0xffff);                         // tile_dim1 (tile_dim2=0)
    g1[5] = row_stride_elems;                          // tensor_dim0_stride[31:0]
    g1[6] = 0;
    g1[7] = 0;
    i32x4 z4 = {0, 0, 0, 0};
#if defined(__clang_major__) && __clang_major__ >= 23
    i32x8 z8 = {0, 0, 0, 0, 0, 0, 0, 0};
    __builtin_amdgcn_tensor_load_to_lds(g0, g1, z4, z4, z8, 0);
#else
    __builtin_amdgcn_tensor_load_to_lds(g0, g1, z4, z4, 0);
#endif
}
#endif // HAVE_TDM

// ---------------------------------------------------------------------------
// 0) Pack grouped-conv weights (OIHW f32) -> [g][m(64)][k(1152)] f16,
//    k = (kh*3+kw)*128 + ci  (matches NHWC activation gather order)
// ---------------------------------------------------------------------------
__global__ __launch_bounds__(256) void pack_weights_k(
    const float* __restrict__ w, _Float16* __restrict__ wpk)
{
    int idx = blockIdx.x * 256 + threadIdx.x;     // 2*64*1152 = 147456 total
    if (idx >= 2 * MTILE * KDIM) return;
    int k  = idx % KDIM;
    int m  = (idx / KDIM) % MTILE;
    int g  = idx / (KDIM * MTILE);
    int ci = k % 128;
    int kk = k / 128;                              // kh*3+kw
    float v = w[(((long)(g * 64 + m) * 128 + ci) * 9) + kk];
    wpk[idx] = (_Float16)v;
}

// ---------------------------------------------------------------------------
// 1) channel_shuffle(groups=2) + sign()  -> f16 NHWC activations
//    shuffled channel c reads input channel (c&1)*128 + (c>>1)
// ---------------------------------------------------------------------------
__global__ __launch_bounds__(256) void binact_shuffle_k(
    const float* __restrict__ x, _Float16* __restrict__ act)
{
    long idx = (long)blockIdx.x * 256 + threadIdx.x;   // NHW*256 threads
    int  c   = (int)(idx & 255);
    long p   = idx >> 8;                               // position in [0,NHW)
    int  n   = (int)(p / BHW);
    int  hw  = (int)(p % BHW);
    int  sc  = (c & 1) * 128 + (c >> 1);
    float v  = x[((long)n * CIO + sc) * BHW + hw];
    float s  = (v > 0.f) ? 1.f : ((v < 0.f) ? -1.f : 0.f);
    act[idx] = (_Float16)s;
}

// ---------------------------------------------------------------------------
// 2) Grouped 3x3 conv as implicit GEMM on WMMA f16 -> f32, LeakyReLU fused.
//    act: [NHW][256] NHWC f16 (signs). wpk: [2][64][1152] f16.
//    out: [N][128][HW] f32.
//    Block = 256 thr (8 wave32). Tile: M=64 (whole group) x N=128 positions.
//    A tile staged via TDM DMA; B tile via async global->LDS (where available).
// ---------------------------------------------------------------------------
__global__ __launch_bounds__(256) void conv_wmma_k(
    const _Float16* __restrict__ act,
    const _Float16* __restrict__ wpk,
    float* __restrict__ out)
{
    __shared__ __align__(32) _Float16 sA[MTILE][KT];   // [m][k]
    __shared__ __align__(32) _Float16 sB[NTILE][KT];   // [n][k] (B transposed)

    const int g       = blockIdx.y;
    const int posBase = blockIdx.x * NTILE;
    const int tid     = threadIdx.x;
    const int lane    = tid & 31;
    const int wave    = tid >> 5;

    // B-tile load assignment: 2 threads per position row (16 f16 each)
    const int jb    = tid >> 1;
    const int koffb = (tid & 1) * 16;
    const int posj  = posBase + jb;
    const int nb    = posj / BHW;
    const int hwb   = posj % BHW;
    const int yj    = hwb / BW;
    const int xj    = hwb % BW;

    // A-tile fallback load assignment: 4 threads per m row (8 f16 each)
    const int ma = tid >> 2;
    const int ca = (tid & 3) * 8;
    const _Float16* wbase = wpk + (long)g * MTILE * KDIM;

    // Launder the sA read pointer: in the TDM path the compiler sees no
    // stores to sA, so keep it from reasoning about the LDS contents.
    const _Float16* sAr = &sA[0][0];
    asm volatile("" : "+v"(sAr)::"memory");

#if HAVE_TDM
    const unsigned ldsA =
        (unsigned)(unsigned long long)(AS3 void*)&sA[0][0];
#endif

    v8f acc[4];
    #pragma unroll
    for (int i = 0; i < 4; ++i)
        #pragma unroll
        for (int e = 0; e < 8; ++e) acc[i][e] = 0.f;

    for (int kt = 0; kt < KITER; ++kt) {
        const int kk  = kt >> 2;            // 0..8 = kh*3+kw
        const int dh  = kk / 3 - 1;
        const int dw  = kk % 3 - 1;
        const int cib = (kt & 3) * 32;      // ci base within the 128 group chans

        // ---- stage A tile (weights 64x32 f16, row stride KDIM) ----
#if HAVE_TDM
        if (wave == 0) {
            tdm_load_tile_f16(ldsA, wbase + kt * KT, KT, MTILE, KDIM);
            wait_tensor_zero();             // TENSORcnt -> 0 before publishing
        }
#else
        *(h8*)&sA[ma][ca] =
            *(const h8*)(wbase + (long)ma * KDIM + kt * KT + ca);
        if (kt + 1 < KITER)
            __builtin_prefetch(wbase + (long)ma * KDIM + (kt + 1) * KT + ca, 0, 1);
#endif

        // ---- stage B tile (binarized activations), padding-aware ----
        {
            int y = yj + dh, x = xj + dw;
            bool valid = ((unsigned)y < BH) && ((unsigned)x < BW);
            const _Float16* src =
                act + (((long)nb * BHW + (valid ? (y * BW + x) : 0)) * CIO)
                    + g * 128 + cib + koffb;
#if HAVE_ASYNC_LDS
            if (valid) {
                __builtin_amdgcn_global_load_async_to_lds_b128(
                    (AS1 i32x4*)src,
                    (AS3 i32x4*)&sB[jb][koffb], 0, 0);
                __builtin_amdgcn_global_load_async_to_lds_b128(
                    (AS1 i32x4*)(src + 8),
                    (AS3 i32x4*)&sB[jb][koffb + 8], 0, 0);
            } else {
                h8 z;
                #pragma unroll
                for (int e = 0; e < 8; ++e) z[e] = (_Float16)0.f;
                *(h8*)&sB[jb][koffb]     = z;
                *(h8*)&sB[jb][koffb + 8] = z;
            }
            wait_async_zero();              // ASYNCcnt -> 0 before barrier
#else
            if (valid) {
                *(h8*)&sB[jb][koffb]     = *(const h8*)(src);
                *(h8*)&sB[jb][koffb + 8] = *(const h8*)(src + 8);
            } else {
                h8 z;
                #pragma unroll
                for (int e = 0; e < 8; ++e) z[e] = (_Float16)0.f;
                *(h8*)&sB[jb][koffb]     = z;
                *(h8*)&sB[jb][koffb + 8] = z;
            }
#endif
        }
        __syncthreads();

        // ---- B fragment: lane l holds N=l&15, K=(l>>4)*8 + e / 16+(l>>4)*8 + e ----
        v16h bf;
        {
            const int nrow = wave * 16 + (lane & 15);
            const int kb   = (lane >> 4) * 8;
            h8 blo = *(const h8*)&sB[nrow][kb];
            h8 bhi = *(const h8*)&sB[nrow][16 + kb];
            #pragma unroll
            for (int e = 0; e < 8; ++e) { bf[e] = blo[e]; bf[e + 8] = bhi[e]; }
        }

        // ---- 4 M-subtiles: A fragment per ISA 16-bit A layout ----
        #pragma unroll
        for (int mi = 0; mi < 4; ++mi) {
            const int mrow = mi * 16 + (lane & 15);
            const int kb   = (lane >> 4) * 8;         // lanes 0-15: K0..7/16..23
            h8 lo = *(const h8*)(sAr + mrow * KT + kb);
            h8 hi = *(const h8*)(sAr + mrow * KT + 16 + kb);
            v16h af;
            #pragma unroll
            for (int e = 0; e < 8; ++e) { af[e] = lo[e]; af[e + 8] = hi[e]; }

            acc[mi] = __builtin_amdgcn_wmma_f32_16x16x32_f16(
                false, af, false, bf, (short)0, acc[mi], false, false);
        }
        __syncthreads();
    }

    // ---- epilogue: LeakyReLU, scatter to NCHW f32 ----
    // C layout: lane l -> N = l&15 (+wave subtile), VGPR v -> M = v + 8*(l>>4)
    const int nidx = wave * 16 + (lane & 15);
    const int pos  = posBase + nidx;
    const int n    = pos / BHW;
    const int hw   = pos % BHW;
    #pragma unroll
    for (int mi = 0; mi < 4; ++mi) {
        #pragma unroll
        for (int v = 0; v < 8; ++v) {
            int   m   = mi * 16 + v + 8 * (lane >> 4);
            float val = acc[mi][v];
            val = (val >= 0.f) ? val : SLOPE * val;
            out[((long)n * CMID + g * 64 + m) * BHW + hw] = val;
        }
    }
}

// ---------------------------------------------------------------------------
// 3) Per-channel batch stats (sum, sumsq) over (N,H,W). grid=(98 chunks, C)
// ---------------------------------------------------------------------------
__global__ __launch_bounds__(256) void bn_stats_k(
    const float* __restrict__ y, float* __restrict__ sums,
    float* __restrict__ sqs, int C)
{
    __shared__ float ssum[256];
    __shared__ float ssq[256];
    const int tid = threadIdx.x;
    const int c   = blockIdx.y;
    const long q0 = (long)blockIdx.x * 1024;
    float s = 0.f, s2 = 0.f;
    #pragma unroll
    for (int i = 0; i < 4; ++i) {
        long q  = q0 + i * 256 + tid;          // 98*1024 == NHW exactly
        int  n  = (int)(q / BHW);
        int  hw = (int)(q % BHW);
        float v = y[((long)n * C + c) * BHW + hw];
        s += v; s2 += v * v;
    }
    ssum[tid] = s; ssq[tid] = s2;
    __syncthreads();
    for (int off = 128; off > 0; off >>= 1) {
        if (tid < off) { ssum[tid] += ssum[tid + off]; ssq[tid] += ssq[tid + off]; }
        __syncthreads();
    }
    if (tid == 0) { atomicAdd(&sums[c], ssum[0]); atomicAdd(&sqs[c], ssq[0]); }
}

// ---------------------------------------------------------------------------
// 4) x3 = concat( BN1(y1) + xa , xb ) ; xa/xb come from shuffled input x
// ---------------------------------------------------------------------------
__global__ __launch_bounds__(256) void build_x3_k(
    const float* __restrict__ x, const float* __restrict__ y1,
    const float* __restrict__ s1, const float* __restrict__ q1,
    const float* __restrict__ g1, const float* __restrict__ b1,
    float* __restrict__ x3)
{
    long idx = (long)blockIdx.x * 256 + threadIdx.x;   // N*256*HW
    int  hw  = (int)(idx % BHW);
    long t   = idx / BHW;
    int  c   = (int)(t % CIO);
    int  n   = (int)(t / CIO);
    int  sc  = (c & 1) * 128 + (c >> 1);               // shuffled source chan
    float xs = x[((long)n * CIO + sc) * BHW + hw];
    float o;
    if (c < 128) {
        float mean = s1[c] * INV_NHW;
        float var  = q1[c] * INV_NHW - mean * mean;
        float v    = y1[((long)n * CMID + c) * BHW + hw];
        float bn   = (v - mean) * rsqrtf(var + EPSV) * g1[c] + b1[c];
        o = bn + xs;
    } else {
        o = xs;
    }
    x3[idx] = o;
}

// ---------------------------------------------------------------------------
// 5) x3s = channel_shuffle(BN2(x3)) f32 ; act2 = sign(x3s) f16 NHWC
// ---------------------------------------------------------------------------
__global__ __launch_bounds__(256) void bn2_shuffle_k(
    const float* __restrict__ x3,
    const float* __restrict__ s2, const float* __restrict__ q2,
    const float* __restrict__ g2, const float* __restrict__ b2,
    float* __restrict__ x3s, _Float16* __restrict__ act2)
{
    long idx = (long)blockIdx.x * 256 + threadIdx.x;   // indexed by OUTPUT chan
    int  hw  = (int)(idx % BHW);
    long t   = idx / BHW;
    int  c   = (int)(t % CIO);
    int  n   = (int)(t / CIO);
    int  sc  = (c & 1) * 128 + (c >> 1);
    float v    = x3[((long)n * CIO + sc) * BHW + hw];
    float mean = s2[sc] * INV_NHW;
    float var  = q2[sc] * INV_NHW - mean * mean;
    float bn   = (v - mean) * rsqrtf(var + EPSV) * g2[sc] + b2[sc];
    x3s[idx] = bn;
    float s = (bn > 0.f) ? 1.f : ((bn < 0.f) ? -1.f : 0.f);
    act2[((long)n * BHW + hw) * CIO + c] = (_Float16)s;
}

// ---------------------------------------------------------------------------
// 6) out = concat( BN3(y4) + x3s[:,:128] , x3s[:,128:] )
// ---------------------------------------------------------------------------
__global__ __launch_bounds__(256) void finalize_k(
    const float* __restrict__ y4, const float* __restrict__ x3s,
    const float* __restrict__ s3, const float* __restrict__ q3,
    const float* __restrict__ g3, const float* __restrict__ b3,
    float* __restrict__ out)
{
    long idx = (long)blockIdx.x * 256 + threadIdx.x;   // N*256*HW
    int  hw  = (int)(idx % BHW);
    long t   = idx / BHW;
    int  c   = (int)(t % CIO);
    int  n   = (int)(t / CIO);
    if (c < 128) {
        float mean = s3[c] * INV_NHW;
        float var  = q3[c] * INV_NHW - mean * mean;
        float v    = y4[((long)n * CMID + c) * BHW + hw];
        float bn   = (v - mean) * rsqrtf(var + EPSV) * g3[c] + b3[c];
        out[idx] = bn + x3s[idx];
    } else {
        out[idx] = x3s[idx];
    }
}

// ---------------------------------------------------------------------------
// launch
// ---------------------------------------------------------------------------
extern "C" void kernel_launch(void* const* d_in, const int* in_sizes, int n_in,
                              void* d_out, int out_size, void* d_ws, size_t ws_size,
                              hipStream_t stream)
{
    const float* x  = (const float*)d_in[0];
    const float* w1 = (const float*)d_in[1];
    const float* w2 = (const float*)d_in[2];
    const float* g1 = (const float*)d_in[3];
    const float* b1 = (const float*)d_in[4];
    const float* g2 = (const float*)d_in[5];
    const float* b2 = (const float*)d_in[6];
    const float* g3 = (const float*)d_in[7];
    const float* b3 = (const float*)d_in[8];
    float* out = (float*)d_out;

    char*  ws  = (char*)d_ws;
    size_t off = 0;
    auto alloc = [&](size_t bytes) -> void* {
        void* p = ws + off;
        off = (off + bytes + 255) & ~(size_t)255;
        return p;
    };

    _Float16* wpk1 = (_Float16*)alloc((size_t)2 * MTILE * KDIM * 2);
    _Float16* wpk2 = (_Float16*)alloc((size_t)2 * MTILE * KDIM * 2);
    _Float16* act1 = (_Float16*)alloc((size_t)BNHW * CIO * 2);
    _Float16* act2 = (_Float16*)alloc((size_t)BNHW * CIO * 2);
    float*    y1   = (float*)alloc((size_t)BNHW * CMID * 4);
    float*    y4   = (float*)alloc((size_t)BNHW * CMID * 4);
    float*    x3   = (float*)alloc((size_t)BNHW * CIO * 4);
    float*    x3s  = (float*)alloc((size_t)BNHW * CIO * 4);
    float*    st   = (float*)alloc(1024 * 4);
    float* s1 = st;       float* q1 = st + 128;
    float* s2 = st + 256; float* q2 = st + 512;
    float* s3 = st + 768; float* q3 = st + 896;

    (void)hipMemsetAsync(st, 0, 1024 * sizeof(float), stream);

    const int elemBlocks = (int)(((long)BNHW * CIO) / 256);   // 100352

    pack_weights_k<<<(2 * MTILE * KDIM + 255) / 256, 256, 0, stream>>>(w1, wpk1);
    pack_weights_k<<<(2 * MTILE * KDIM + 255) / 256, 256, 0, stream>>>(w2, wpk2);

    binact_shuffle_k<<<elemBlocks, 256, 0, stream>>>(x, act1);
    conv_wmma_k<<<dim3(BNHW / NTILE, 2), 256, 0, stream>>>(act1, wpk1, y1);
    bn_stats_k<<<dim3(98, CMID), 256, 0, stream>>>(y1, s1, q1, CMID);
    build_x3_k<<<elemBlocks, 256, 0, stream>>>(x, y1, s1, q1, g1, b1, x3);
    bn_stats_k<<<dim3(98, CIO), 256, 0, stream>>>(x3, s2, q2, CIO);
    bn2_shuffle_k<<<elemBlocks, 256, 0, stream>>>(x3, s2, q2, g2, b2, x3s, act2);
    conv_wmma_k<<<dim3(BNHW / NTILE, 2), 256, 0, stream>>>(act2, wpk2, y4);
    bn_stats_k<<<dim3(98, CMID), 256, 0, stream>>>(y4, s3, q3, CMID);
    finalize_k<<<elemBlocks, 256, 0, stream>>>(y4, x3s, s3, q3, g3, b3, out);
}
